// DifferentialCrossAttentionHead_83021717832617
// MI455X (gfx1250) — matrix operand order repaired and checked by reading, not compiled, not run on hardware
//
#include <hip/hip_runtime.h>
#include <hip/hip_bf16.h>

typedef __attribute__((ext_vector_type(16))) _Float16 v16h;
typedef __attribute__((ext_vector_type(8)))  float    v8f;
typedef __attribute__((ext_vector_type(4)))  unsigned v4u;

#define NB 4
#define TQ 2048
#define TK 2048
#define ED 1024
#define HD 128

__device__ __forceinline__ v8f wmma16(v16h a, v16h b, v8f c) {
  // D = A(16x32 f16) * B(32x16 f16) + C(16x16 f32)
  return __builtin_amdgcn_wmma_f32_16x16x32_f16(false, a, false, b, (short)0, c,
                                                false, false);
}

// A fragment: lane holds row (lane&15), K runs {half*8..+7} and {16+half*8..+7}.
__device__ __forceinline__ v16h ldA(const _Float16* rowbase, int half) {
  union { v16h h; v4u q[2]; } f;
  f.q[0] = *(const v4u*)(rowbase + half * 8);
  f.q[1] = *(const v4u*)(rowbase + 16 + half * 8);
  return f.h;
}
// B fragment: lane holds col (lane&15), K runs {half*8..+7} and {16+half*8..+7}.
__device__ __forceinline__ v16h ldB(const _Float16* runbase, int half) {
  union { v16h h; v4u q[2]; } f;
  f.q[0] = *(const v4u*)(runbase + half * 8);
  f.q[1] = *(const v4u*)(runbase + 16 + half * 8);
  return f.h;
}

// ---------------------------------------------------------------------------
// lambda = exp(lq1.lk1) - exp(lq2.lk2) + lambda_init   (one wave)
// ---------------------------------------------------------------------------
__global__ void lam_kernel(const float* __restrict__ lq1, const float* __restrict__ lk1,
                           const float* __restrict__ lq2, const float* __restrict__ lk2,
                           const float* __restrict__ linit, float* __restrict__ lam) {
  int lane = threadIdx.x;
  float d1 = 0.f, d2 = 0.f;
  for (int i = lane; i < HD; i += 32) {
    d1 += lq1[i] * lk1[i];
    d2 += lq2[i] * lk2[i];
  }
  #pragma unroll
  for (int off = 16; off > 0; off >>= 1) {
    d1 += __shfl_xor(d1, off);
    d2 += __shfl_xor(d2, off);
  }
  if (lane == 0) *lam = __expf(d1) - __expf(d2) + linit[0];
}

// ---------------------------------------------------------------------------
// Projection: dst(f16) = src(8192x1024 f32) @ W(1024x128) + bias
// blockIdx.z: 0 -> q from x, 1 -> k from enc, 2 -> v from enc
// ---------------------------------------------------------------------------
__global__ __launch_bounds__(256) void proj_kernel(
    const float* __restrict__ x, const float* __restrict__ enc,
    const float* __restrict__ Wq, const float* __restrict__ bq,
    const float* __restrict__ Wk, const float* __restrict__ bk,
    const float* __restrict__ Wv, const float* __restrict__ bv,
    _Float16* __restrict__ qh, _Float16* __restrict__ kh, _Float16* __restrict__ vh) {
  const float* src; const float* W; const float* bias; _Float16* dst;
  if (blockIdx.z == 0)      { src = x;   W = Wq; bias = bq; dst = qh; }
  else if (blockIdx.z == 1) { src = enc; W = Wk; bias = bk; dst = kh; }
  else                      { src = enc; W = Wv; bias = bv; dst = vh; }

  const int tid  = threadIdx.x;
  const int wave = tid >> 5;
  const int lane = tid & 31;
  const int half = lane >> 4;
  const int mn   = lane & 15;
  const int rowblk = blockIdx.x * 128;

  __shared__ _Float16 xst[128 * 32];   // 8 KB: 128 rows x 32 k (row-major)
  __shared__ _Float16 wstT[HD * 32];   // 8 KB: 128 cols x 32 k (transposed)

  const int xr = tid >> 1;             // x staging: row 0..127
  const int xc = (tid & 1) * 16;       // col 0 or 16
  const int wr0 = (tid >> 4) * 2;      // W staging: rows wr0..wr0+1
  const int wc0 = (tid & 15) * 8;      // cols wc0..wc0+7

  v8f acc[8];
  #pragma unroll
  for (int nt = 0; nt < 8; ++nt)
    #pragma unroll
    for (int i = 0; i < 8; ++i) acc[nt][i] = 0.f;

  for (int kc = 0; kc < ED; kc += 32) {
    __syncthreads();
    {  // x: coalesced f32 reads, convert, two b128 LDS stores
      const float* xs = src + (size_t)(rowblk + xr) * ED + kc + xc;
      union { _Float16 h[8]; v4u q; } u0, u1;
      #pragma unroll
      for (int j = 0; j < 8; ++j) { u0.h[j] = (_Float16)xs[j]; u1.h[j] = (_Float16)xs[8 + j]; }
      *(v4u*)(xst + xr * 32 + xc)     = u0.q;
      *(v4u*)(xst + xr * 32 + xc + 8) = u1.q;
    }
    {  // W: read 2 rows x 8 cols, pack row-pairs, b32 transposed stores
      const float* w0 = W + (size_t)(kc + wr0) * HD + wc0;
      const float* w1 = w0 + HD;
      #pragma unroll
      for (int j = 0; j < 8; ++j) {
        union { _Float16 h[2]; unsigned u; } p;
        p.h[0] = (_Float16)w0[j];
        p.h[1] = (_Float16)w1[j];
        *(unsigned*)(wstT + (wc0 + j) * 32 + wr0) = p.u;
      }
    }
    __syncthreads();

    // preload all fragments so DS loads pipeline ahead of the WMMA chain
    v16h a = ldA(xst + (wave * 16 + mn) * 32, half);
    v16h bfr[8];
    #pragma unroll
    for (int nt = 0; nt < 8; ++nt)
      bfr[nt] = ldB(wstT + (nt * 16 + mn) * 32, half);
    #pragma unroll
    for (int nt = 0; nt < 8; ++nt)
      acc[nt] = wmma16(a, bfr[nt], acc[nt]);
  }

  const int row0 = rowblk + wave * 16;
  #pragma unroll
  for (int nt = 0; nt < 8; ++nt) {
    float bval = bias[nt * 16 + mn];
    #pragma unroll
    for (int r = 0; r < 8; ++r) {
      float val = acc[nt][r] + bval;
      dst[(size_t)(row0 + r + 8 * half) * HD + nt * 16 + mn] = (_Float16)val;
    }
  }
}

// ---------------------------------------------------------------------------
// Fused differential flash-attention: per block 128 q-rows, 64-key tiles.
// K tile staged via 4x global_load_async_to_lds_b128 per thread (ASYNCcnt);
// V tile staged transposed with b64-packed stores; dual online softmax.
// ---------------------------------------------------------------------------
__global__ __launch_bounds__(256) void attn_kernel(
    const _Float16* __restrict__ qh, const _Float16* __restrict__ kh,
    const _Float16* __restrict__ vh, const float* __restrict__ lamp,
    float* __restrict__ out) {
  const int bb   = blockIdx.y;
  const int tid  = threadIdx.x;
  const int wave = tid >> 5;
  const int lane = tid & 31;
  const int half = lane >> 4;
  const int mn   = lane & 15;
  const int row0 = blockIdx.x * 128 + wave * 16;

  const _Float16* Q = qh + (size_t)bb * TQ * HD;
  const _Float16* K = kh + (size_t)bb * TK * HD;
  const _Float16* V = vh + (size_t)bb * TK * HD;
  const float lam   = *lamp;
  const float scale = 0.125f;  // 1/sqrt(64)

  __shared__ _Float16 kst[64 * HD];          // 16 KB: 64 keys x 128 dims (row-major)
  __shared__ _Float16 vstT[HD * 64];         // 16 KB: 128 cols x 64 keys (transposed)
  __shared__ _Float16 pst[8 * 2 * 16 * 64];  // 32 KB: per-wave P1/P2 (16x64) tiles
  _Float16* p1t = pst + wave * 2048;
  _Float16* p2t = p1t + 1024;

  // K tile async staging: 64 B per thread
  const unsigned koff = (unsigned)(unsigned long long)(const void*)kst + (unsigned)tid * 64u;
  const unsigned long long kgbase =
      (unsigned long long)(const void*)K + (unsigned long long)tid * 64ull;
  // V tile staging: 4 rows x 8 cols per thread
  const int vr0 = (tid >> 4) * 4;   // 0..60
  const int vc0 = (tid & 15) * 8;   // 0..120

  // Q fragments: c=0,1 -> first 64 dims (q1); c=2,3 -> last 64 dims (q2)
  v16h qa[4];
  #pragma unroll
  for (int c = 0; c < 4; ++c)
    qa[c] = ldA(Q + (size_t)(row0 + mn) * HD + c * 32, half);

  v8f o1[8], o2[8];
  float m1[8], l1[8], m2[8], l2[8];
  #pragma unroll
  for (int nt = 0; nt < 8; ++nt)
    #pragma unroll
    for (int i = 0; i < 8; ++i) { o1[nt][i] = 0.f; o2[nt][i] = 0.f; }
  #pragma unroll
  for (int r = 0; r < 8; ++r) { m1[r] = -1e30f; l1[r] = 0.f; m2[r] = -1e30f; l2[r] = 0.f; }

  for (int kb = 0; kb < TK; kb += 64) {
    __syncthreads();  // previous tile fully consumed by all waves

    // ---- K tile: async DMA to LDS ----
    {
      unsigned long long gk = kgbase + (unsigned long long)kb * (HD * 2);
      asm volatile("global_load_async_to_lds_b128 %0, %1, off"
                   :: "v"(koff), "v"(gk) : "memory");
      asm volatile("global_load_async_to_lds_b128 %0, %1, off offset:16"
                   :: "v"(koff), "v"(gk) : "memory");
      asm volatile("global_load_async_to_lds_b128 %0, %1, off offset:32"
                   :: "v"(koff), "v"(gk) : "memory");
      asm volatile("global_load_async_to_lds_b128 %0, %1, off offset:48"
                   :: "v"(koff), "v"(gk) : "memory");
    }
    // ---- V tile: b128 reads, b64-packed transposed stores ----
    {
      const _Float16* vsrc = V + (size_t)(kb + vr0) * HD + vc0;
      union { _Float16 h[8]; v4u q; } rowd[4];
      #pragma unroll
      for (int rr = 0; rr < 4; ++rr)
        rowd[rr].q = *(const v4u*)(vsrc + (size_t)rr * HD);
      #pragma unroll
      for (int j = 0; j < 8; ++j) {
        union { _Float16 h[4]; unsigned long long u; } p;
        p.h[0] = rowd[0].h[j]; p.h[1] = rowd[1].h[j];
        p.h[2] = rowd[2].h[j]; p.h[3] = rowd[3].h[j];
        *(unsigned long long*)(vstT + (size_t)(vc0 + j) * 64 + vr0) = p.u;
      }
    }
    asm volatile("s_wait_asynccnt 0" ::: "memory");
    __syncthreads();

    // ---- scores: four 16-col subtiles; 4 fragments preloaded per subtile ----
    v8f s1[4], s2[4];
    #pragma unroll
    for (int t = 0; t < 4; ++t) {
      const _Float16* kp = kst + (size_t)(t * 16 + mn) * HD;
      v16h f10 = ldB(kp, half);
      v16h f11 = ldB(kp + 32, half);
      v16h f20 = ldB(kp + 64, half);
      v16h f21 = ldB(kp + 96, half);
      #pragma unroll
      for (int i = 0; i < 8; ++i) { s1[t][i] = 0.f; s2[t][i] = 0.f; }
      s1[t] = wmma16(qa[0], f10, s1[t]);
      s1[t] = wmma16(qa[1], f11, s1[t]);
      s2[t] = wmma16(qa[2], f20, s2[t]);
      s2[t] = wmma16(qa[3], f21, s2[t]);
    }

    // ---- dual online softmax over 64 keys; write P tiles ----
    #pragma unroll
    for (int r = 0; r < 8; ++r) {
      float a_[4], b_[4];
      #pragma unroll
      for (int t = 0; t < 4; ++t) { a_[t] = s1[t][r] * scale; b_[t] = s2[t][r] * scale; }
      float t1 = fmaxf(fmaxf(a_[0], a_[1]), fmaxf(a_[2], a_[3]));
      float t2 = fmaxf(fmaxf(b_[0], b_[1]), fmaxf(b_[2], b_[3]));
      #pragma unroll
      for (int off = 1; off < 16; off <<= 1) {  // reduce across the 16 lanes of a row
        t1 = fmaxf(t1, __shfl_xor(t1, off));
        t2 = fmaxf(t2, __shfl_xor(t2, off));
      }
      float nm1 = fmaxf(m1[r], t1), nm2 = fmaxf(m2[r], t2);
      float c1 = __expf(m1[r] - nm1), c2 = __expf(m2[r] - nm2);
      m1[r] = nm1; m2[r] = nm2;
      float ps1 = 0.f, ps2 = 0.f;
      int prow = (r + 8 * half) * 64;
      #pragma unroll
      for (int t = 0; t < 4; ++t) {
        float pv1 = __expf(a_[t] - nm1); ps1 += pv1;
        float pv2 = __expf(b_[t] - nm2); ps2 += pv2;
        p1t[prow + t * 16 + mn] = (_Float16)pv1;
        p2t[prow + t * 16 + mn] = (_Float16)pv2;
      }
      l1[r] = l1[r] * c1 + ps1;                 // per-lane partial row sums
      l2[r] = l2[r] * c2 + ps2;
      #pragma unroll
      for (int nt = 0; nt < 8; ++nt) { o1[nt][r] *= c1; o2[nt][r] *= c2; }
    }

    // ---- O += P @ V, rolling V-fragment double buffer ----
    #pragma unroll
    for (int c = 0; c < 2; ++c) {
      v16h pa1 = ldA(p1t + mn * 64 + c * 32, half);
      v16h pa2 = ldA(p2t + mn * 64 + c * 32, half);
      v16h vb_c = ldB(vstT + (size_t)mn * 64 + c * 32, half);  // nt = 0
      #pragma unroll
      for (int nt = 0; nt < 8; ++nt) {
        v16h vb_n = (nt < 7)
            ? ldB(vstT + (size_t)((nt + 1) * 16 + mn) * 64 + c * 32, half)
            : vb_c;
        o1[nt] = wmma16(pa1, vb_c, o1[nt]);
        o2[nt] = wmma16(pa2, vb_c, o2[nt]);
        vb_c = vb_n;
      }
    }
  }

  // finalize: reduce row sums across lanes, normalize, combine, store
  #pragma unroll
  for (int r = 0; r < 8; ++r) {
    #pragma unroll
    for (int off = 1; off < 16; off <<= 1) {
      l1[r] += __shfl_xor(l1[r], off);
      l2[r] += __shfl_xor(l2[r], off);
    }
  }
  float* outp = out + (size_t)bb * TQ * HD;
  #pragma unroll
  for (int nt = 0; nt < 8; ++nt)
    #pragma unroll
    for (int r = 0; r < 8; ++r) {
      float val = o1[nt][r] / l1[r] - lam * (o2[nt][r] / l2[r]);
      outp[(size_t)(row0 + r + 8 * half) * HD + nt * 16 + mn] = val;
    }
}

// ---------------------------------------------------------------------------
extern "C" void kernel_launch(void* const* d_in, const int* in_sizes, int n_in,
                              void* d_out, int out_size, void* d_ws, size_t ws_size,
                              hipStream_t stream) {
  (void)in_sizes; (void)n_in; (void)out_size; (void)ws_size;
  const float* x    = (const float*)d_in[0];
  const float* enc  = (const float*)d_in[1];
  const float* Wq   = (const float*)d_in[2];
  const float* bq   = (const float*)d_in[3];
  const float* Wk   = (const float*)d_in[4];
  const float* bk   = (const float*)d_in[5];
  const float* Wv   = (const float*)d_in[6];
  const float* bv   = (const float*)d_in[7];
  const float* lq1  = (const float*)d_in[8];
  const float* lk1  = (const float*)d_in[9];
  const float* lq2  = (const float*)d_in[10];
  const float* lk2  = (const float*)d_in[11];
  const float* lini = (const float*)d_in[12];

  char* ws = (char*)d_ws;
  _Float16* qh  = (_Float16*)(ws + (size_t)0);
  _Float16* kh  = (_Float16*)(ws + (size_t)2 * 1024 * 1024);
  _Float16* vh  = (_Float16*)(ws + (size_t)4 * 1024 * 1024);
  float*    lam = (float*)(ws + (size_t)6 * 1024 * 1024);

  lam_kernel<<<1, 32, 0, stream>>>(lq1, lk1, lq2, lk2, lini, lam);

  dim3 pg((NB * TQ) / 128, 1, 3);   // 64 x 1 x 3
  proj_kernel<<<pg, 256, 0, stream>>>(x, enc, Wq, bq, Wk, bk, Wv, bv, qh, kh, vh);

  dim3 ag(TQ / 128, NB, 1);         // 16 x 4
  attn_kernel<<<ag, 256, 0, stream>>>(qh, kh, vh, lam, (float*)d_out);
}
